// THLSTM_69758858821995
// MI455X (gfx1250) — matrix operand order repaired and checked by reading, not compiled, not run on hardware
//
#include <hip/hip_runtime.h>
#include <hip/hip_bf16.h>
#include <cstddef>

#define TH_B   1024
#define TH_S   256
#define TH_F   64
#define TH_HD  64
#define TH_MEM 10
#define ROWS   16     // batch rows per workgroup (one WMMA M tile)
#define NTH    128    // 4 waves of 32

// ---------------- LDS layout (bytes) --------------------------------------
#define OFF_WG0P   0                      // 96x320 u32  bf16-pair packed Wg0
#define OFF_WG1P   122880                 // 96x320 u32
#define OFF_WS0P   245760                 // 64x64  u32  rows 0..127 of Ws0
#define OFF_WS1P   262144                 // 64x64  u32
#define OFF_A0     278528                 // 16x96  u32  [h0 | x | s0] bf16 pairs
#define OFF_A1     284672                 // 16x96  u32  [h1 | h0 | s1]
#define OFF_XB     290816                 // 16x64  f32  transient s/h export
#define OFF_HIST   294912                 // 10x16x32 u32 packed h1 history
#define OFF_DLS    315392                 // 16 f32 time deltas
#define OFF_BS0    315456
#define OFF_BS1    (OFF_BS0 + 256)
#define OFF_BG0    (OFF_BS1 + 256)
#define OFF_BG1    (OFF_BG0 + 1280)
#define OFF_WSL0   (OFF_BG1 + 1280)       // Ws0 row 128 (delta column)
#define OFF_WSL1   (OFF_WSL0 + 256)
#define OFF_XS0    (OFF_WSL1 + 256)       // 16x64 f32 TDM stage buffer 0
#define OFF_XS1    (OFF_XS0 + 4096)       // 16x64 f32 TDM stage buffer 1
#define OFF_TPREV  (OFF_XS1 + 4096)       // 16 f32 previous time values
#define SMEM_BYTES (OFF_TPREV + 64)       // 327296 <= 327680 (320KB WGP LDS)

#if defined(__has_builtin)
#if __has_builtin(__builtin_amdgcn_tensor_load_to_lds) && \
    __has_builtin(__builtin_amdgcn_s_wait_tensorcnt)
#define HAVE_TDM 1
#endif
#endif

typedef __attribute__((ext_vector_type(16))) __bf16 bf16x16;
typedef __attribute__((ext_vector_type(8)))  float  f32x8;
typedef __attribute__((ext_vector_type(4)))  unsigned u32x4;
typedef __attribute__((ext_vector_type(4)))  int    i32x4;
typedef __attribute__((ext_vector_type(8)))  int    i32x8;

union Frag { unsigned u[8]; bf16x16 b; };

__device__ __forceinline__ unsigned pack2(float a, float b) {
  unsigned ua = __float_as_uint(a), ub = __float_as_uint(b);
  ua = (ua + 0x7FFFu + ((ua >> 16) & 1u)) >> 16;   // RNE f32 -> bf16
  ub = (ub + 0x7FFFu + ((ub >> 16) & 1u)) >> 16;
  return (ub << 16) | (ua & 0xFFFFu);
}
__device__ __forceinline__ float bflo(unsigned u) { return __uint_as_float(u << 16); }
__device__ __forceinline__ float bfhi(unsigned u) { return __uint_as_float(u & 0xFFFF0000u); }

// Branch-free transcendentals on the HW trans units (v_exp_f32 / v_rcp_f32).
// tanh(x) = 1 - 2/(exp(2x)+1): saturates to +-1 for large |x|, no exec-mask churn.
__device__ __forceinline__ float rcpf(float v)  { return __builtin_amdgcn_rcpf(v); }
__device__ __forceinline__ float sigf(float v)  { return rcpf(1.f + __expf(-v)); }
__device__ __forceinline__ float tanhf_hw(float v) {
  return 1.f - 2.f * rcpf(__expf(2.f * v) + 1.f);
}

// A fragment: 16x32 bf16 tile, M = lane&15, K-half selected by lane>>4.
// A-buffer rows hold bf16 pairs: pair p <-> K = {2p, 2p+1}, row stride 96.
__device__ __forceinline__ void loadA(Frag& a, const unsigned* Ab, int kt, int ln, int lg) {
  const unsigned* row = Ab + ln * 96 + kt * 16;
  #pragma unroll
  for (int v = 0; v < 4; ++v) {
    a.u[v]     = row[v + 4 * lg];          // K = 2v+8g .. +1
    a.u[4 + v] = row[8 + v + 4 * lg];      // K = 16+2v+8g .. +1
  }
}
// B fragment: 32x16 bf16 tile. N = lane&15; VGPR v holds K pair (v + 4g) for v<4, (8 + v-4 + 4g) for v>=4.
__device__ __forceinline__ void loadB(Frag& b, const unsigned* Wp, int ncols,
                                      int kt, int nt, int ln, int lg) {
  const unsigned* base = Wp + (kt * 16 + 4 * lg) * ncols + nt * 16 + ln;
  #pragma unroll
  for (int v = 0; v < 8; ++v) b.u[v] = base[(v + (v & 4)) * ncols];
}
__device__ __forceinline__ f32x8 wmma_bf16(const Frag& a, const Frag& b, f32x8 c) {
  return __builtin_amdgcn_wmma_f32_16x16x32_bf16(false, a.b, false, b.b,
                                                 (short)0, c, false, false);
}

#ifdef HAVE_TDM
// Tensor Data Mover: DMA a 16x64 f32 tile (rows strided by S*F elements in
// global) into a contiguous LDS stage buffer.  D# per CDNA5 ISA ch.8.
// This toolchain exposes the 6-arg builtin: (g0, g1, g2, g3, g4, cpol).
__device__ __forceinline__ void issue_x_tdm(const float* gaddr, unsigned lds_off) {
  unsigned long long ga = (unsigned long long)(const void*)gaddr;
  u32x4 g0;
  g0[0] = 1u;                                              // count=1, user mode
  g0[1] = lds_off;                                         // lds_addr (bytes)
  g0[2] = (unsigned)(ga & 0xFFFFFFFFull);                  // global_addr[31:0]
  g0[3] = (unsigned)((ga >> 32) & 0x1FFFFFFull) | (2u << 30); // addr[56:32]|type=2
  i32x8 g1;
  g1[0] = (int)(2u << 16);        // data_size=4B; no pad/iterate/multicast
  g1[1] = (int)(64u << 16);       // tensor_dim0 = 64 elements
  g1[2] = (int)(16u << 16);       // tensor_dim1 = 16 rows
  g1[3] = (int)(64u << 16);       // tile_dim0 = 64
  g1[4] = 16;                     // tile_dim1 = 16
  g1[5] = TH_S * TH_F;            // tensor_dim0_stride = 16384 elements
  g1[6] = 0;
  g1[7] = 0;
  i32x4 z4 = {0, 0, 0, 0};        // 2D tensor: higher-dim groups unused
  i32x8 z8 = {0, 0, 0, 0, 0, 0, 0, 0};
  __builtin_amdgcn_tensor_load_to_lds(g0, g1, z4, z4, z8, 0);
}
#endif

// One LSTM cell step for all 16 rows. Wave `wv` owns columns 16*wv..16*wv+15.
__device__ __forceinline__ void cell_step(unsigned* Ab, const unsigned* WsP,
                                          const unsigned* WgP, const float* bsL,
                                          const float* bgL, const float* wsLast,
                                          const float* dls, float* xb, f32x8& creg,
                                          int wv, int ln, int lg) {
  const int scol = wv * 16 + ln;
  Frag a, b;
  // ---- GEMM1: s = tanh([h,x,d] @ Ws + bs);  d column folded into C init ----
  f32x8 sacc;
  #pragma unroll
  for (int r = 0; r < 8; ++r)
    sacc[r] = bsL[scol] + dls[r + 8 * lg] * wsLast[scol];
  #pragma unroll
  for (int kt = 0; kt < 4; ++kt) {
    loadA(a, Ab, kt, ln, lg);
    loadB(b, WsP, 64, kt, wv, ln, lg);
    sacc = wmma_bf16(a, b, sacc);
  }
  f32x8 sv;
  #pragma unroll
  for (int r = 0; r < 8; ++r) sv[r] = tanhf_hw(sacc[r]);
  // export s (f32) and repack bf16 pairs into A-buffer cols 64..95
  #pragma unroll
  for (int r = 0; r < 8; ++r) xb[(r + 8 * lg) * 64 + scol] = sv[r];
  __syncthreads();
  {
    const int tid = threadIdx.x;
    #pragma unroll
    for (int q = 0; q < 4; ++q) {
      int idx = tid + q * NTH, m = idx >> 5, p = idx & 31;
      Ab[m * 96 + 64 + p] = pack2(xb[m * 64 + 2 * p], xb[m * 64 + 2 * p + 1]);
    }
  }
  __syncthreads();
  // ---- GEMM2: g = [h,x,s] @ Wg + bg  (5 gate N-tiles per wave) ----
  f32x8 gacc[5];
  #pragma unroll
  for (int G = 0; G < 5; ++G)
    #pragma unroll
    for (int r = 0; r < 8; ++r) gacc[G][r] = bgL[G * 64 + scol];
  #pragma unroll
  for (int kt = 0; kt < 6; ++kt) {
    loadA(a, Ab, kt, ln, lg);
    #pragma unroll
    for (int G = 0; G < 5; ++G) {
      loadB(b, WgP, 320, kt, G * 4 + wv, ln, lg);
      gacc[G] = wmma_bf16(a, b, gacc[G]);
    }
  }
  // ---- gates, state update: all in registers (wave owns its 16 columns) ----
  #pragma unroll
  for (int r = 0; r < 8; ++r) {
    float fo = sigf(gacc[0][r]);
    float io = sigf(gacc[1][r]);
    float To = sigf(gacc[2][r]);
    float uo = tanhf_hw(gacc[3][r]);
    float oo = sigf(gacc[4][r]);
    float cn = fo * creg[r] + io * uo + To * sv[r];
    creg[r] = cn;
    xb[(r + 8 * lg) * 64 + scol] = oo * tanhf_hw(cn);   // h_new (f32)
  }
  __syncthreads();
}

extern "C" __global__ __launch_bounds__(NTH)
void thlstm_wmma_kernel(const float* __restrict__ x,
                        const float* __restrict__ Ws0, const float* __restrict__ bs0,
                        const float* __restrict__ Wg0, const float* __restrict__ bg0,
                        const float* __restrict__ Ws1, const float* __restrict__ bs1,
                        const float* __restrict__ Wg1, const float* __restrict__ bg1,
                        const float* __restrict__ Waq, const float* __restrict__ baq,
                        const float* __restrict__ Wah, const float* __restrict__ bah,
                        const float* __restrict__ bias_a, const float* __restrict__ vvec,
                        const float* __restrict__ Wint, const float* __restrict__ bint,
                        float* __restrict__ out) {
  extern __shared__ __align__(16) unsigned char smem[];
  unsigned* Wg0p = (unsigned*)(smem + OFF_WG0P);
  unsigned* Wg1p = (unsigned*)(smem + OFF_WG1P);
  unsigned* Ws0p = (unsigned*)(smem + OFF_WS0P);
  unsigned* Ws1p = (unsigned*)(smem + OFF_WS1P);
  unsigned* A0   = (unsigned*)(smem + OFF_A0);
  unsigned* A1   = (unsigned*)(smem + OFF_A1);
  float*    xb   = (float*)   (smem + OFF_XB);
  unsigned* hist = (unsigned*)(smem + OFF_HIST);
  float*    dls  = (float*)   (smem + OFF_DLS);
  float*    bs0L = (float*)   (smem + OFF_BS0);
  float*    bs1L = (float*)   (smem + OFF_BS1);
  float*    bg0L = (float*)   (smem + OFF_BG0);
  float*    bg1L = (float*)   (smem + OFF_BG1);
  float*    wsl0 = (float*)   (smem + OFF_WSL0);
  float*    wsl1 = (float*)   (smem + OFF_WSL1);
  float*    tprv = (float*)   (smem + OFF_TPREV);

  const int tid = threadIdx.x;
  const int wv  = tid >> 5;
  const int ln  = tid & 15;
  const int lg  = (tid >> 4) & 1;
  const int b0  = blockIdx.x * ROWS;

  // ---- one-time: stage weights into LDS as bf16 K-pair packed tiles ----
  for (int i = tid; i < 64 * 64; i += NTH) {
    int p = i >> 6, n = i & 63;
    Ws0p[i] = pack2(Ws0[(2 * p) * 64 + n], Ws0[(2 * p + 1) * 64 + n]);
    Ws1p[i] = pack2(Ws1[(2 * p) * 64 + n], Ws1[(2 * p + 1) * 64 + n]);
  }
  for (int i = tid; i < 96 * 320; i += NTH) {
    int p = i / 320, n = i % 320;
    Wg0p[i] = pack2(Wg0[(2 * p) * 320 + n], Wg0[(2 * p + 1) * 320 + n]);
    Wg1p[i] = pack2(Wg1[(2 * p) * 320 + n], Wg1[(2 * p + 1) * 320 + n]);
  }
  if (tid < 64) {
    bs0L[tid] = bs0[tid]; bs1L[tid] = bs1[tid];
    wsl0[tid] = Ws0[128 * 64 + tid]; wsl1[tid] = Ws1[128 * 64 + tid];
  }
  for (int i = tid; i < 320; i += NTH) { bg0L[i] = bg0[i]; bg1L[i] = bg1[i]; }
  for (int i = tid; i < 16 * 32; i += NTH) {          // h parts start at zero
    int m = i >> 5, p = i & 31;
    A0[m * 96 + p] = 0u; A1[m * 96 + p] = 0u;
  }
  __syncthreads();

#ifdef HAVE_TDM
  if (wv == 0)   // DMA the t=0 x-tile into stage buffer 0
    issue_x_tdm(x + (size_t)b0 * TH_S * TH_F, OFF_XS0);
#endif

  f32x8 c0, c1;
  #pragma unroll
  for (int r = 0; r < 8; ++r) { c0[r] = 0.f; c1[r] = 0.f; }

  // ---------------- sequential scan over time ----------------
  for (int t = 0; t < TH_S; ++t) {
#ifdef HAVE_TDM
    if (wv == 0) {
      __builtin_amdgcn_s_wait_tensorcnt(0);        // x_t tile landed in LDS
      if (t + 1 < TH_S)                            // DMA t+1 overlaps compute
        issue_x_tdm(x + ((size_t)b0 * TH_S + (t + 1)) * TH_F,
                    (t & 1) ? OFF_XS0 : OFF_XS1);
    }
    __syncthreads();
    {
      const float* xs = (const float*)(smem + ((t & 1) ? OFF_XS1 : OFF_XS0));
      #pragma unroll
      for (int q = 0; q < 4; ++q) {                // bf16-pack into A0 cols 32..63
        int idx = tid + q * NTH, m = idx >> 5, p = idx & 31;
        A0[m * 96 + 32 + p] = pack2(xs[m * 64 + 2 * p], xs[m * 64 + 2 * p + 1]);
        if (p == 0) {                              // time-delta from staged tile
          float tm = xs[m * 64];
          dls[m] = (t > 0) ? (tm - tprv[m]) : 0.f;
          tprv[m] = tm;
        }
      }
    }
#else
    #pragma unroll
    for (int q = 0; q < 4; ++q) {
      int idx = tid + q * NTH, m = idx >> 5, p = idx & 31;
      const float* xp = x + ((size_t)(b0 + m) * TH_S + t) * TH_F + 2 * p;
      A0[m * 96 + 32 + p] = pack2(xp[0], xp[1]);
      if (t + 1 < TH_S) __builtin_prefetch(xp + TH_F, 0, 0);
    }
    if (tid < ROWS) {
      const float* xr = x + ((size_t)(b0 + tid) * TH_S + t) * TH_F;
      dls[tid] = (t > 0) ? (xr[0] - xr[-TH_F]) : 0.f;
    }
#endif
    __syncthreads();

    // layer 0
    cell_step(A0, Ws0p, Wg0p, bs0L, bg0L, wsl0, dls, xb, c0, wv, ln, lg);
    #pragma unroll
    for (int q = 0; q < 4; ++q) {     // h0 -> A0 h-part (next t) + A1 x-part
      int idx = tid + q * NTH, m = idx >> 5, p = idx & 31;
      unsigned u = pack2(xb[m * 64 + 2 * p], xb[m * 64 + 2 * p + 1]);
      A0[m * 96 + p] = u; A1[m * 96 + 32 + p] = u;
    }
    __syncthreads();

    // layer 1
    cell_step(A1, Ws1p, Wg1p, bs1L, bg1L, wsl1, dls, xb, c1, wv, ln, lg);
    const int slot = t - (TH_S - TH_MEM);
    #pragma unroll
    for (int q = 0; q < 4; ++q) {     // h1 -> A1 h-part (+ history ring)
      int idx = tid + q * NTH, m = idx >> 5, p = idx & 31;
      unsigned u = pack2(xb[m * 64 + 2 * p], xb[m * 64 + 2 * p + 1]);
      A1[m * 96 + p] = u;
      if (slot >= 0) hist[(slot * 16 + m) * 32 + p] = u;
    }
    __syncthreads();
  }

  // ---------------- attention epilogue (tiny; VALU f32) ----------------
  float* cb     = (float*)A1;          // A-buffers are dead: reuse as scratch
  float* qproj  = (float*)A0;          // 16x64
  float* scores = (float*)A0 + 1024;   // 16x10
  float* ebuf   = (float*)A0;          // reused after scores computed

  {
    const int scol = wv * 16 + ln;
    #pragma unroll
    for (int r = 0; r < 8; ++r) cb[(r + 8 * lg) * 64 + scol] = c1[r];
  }
  __syncthreads();                     // xb holds final h1 (f32)

  for (int o = tid; o < 16 * 64; o += NTH) {         // q_proj = [h1,c1]@Waq+baq
    int m = o >> 6, hc = o & 63;
    float acc = baq[hc];
    for (int d = 0; d < 64; ++d)
      acc += xb[m * 64 + d] * Waq[d * 64 + hc] + cb[m * 64 + d] * Waq[(64 + d) * 64 + hc];
    qproj[o] = acc;
  }
  __syncthreads();

  for (int it = 0; it < 2; ++it) {                   // scores[m][k]
    int idx = tid + it * NTH;
    if (idx < 16 * TH_MEM) {
      int m = idx / TH_MEM, k = idx % TH_MEM;
      const unsigned* hrow = hist + (k * 16 + m) * 32;
      float sc = 0.f;
      for (int hc = 0; hc < 64; ++hc) {
        float hp = bah[hc];
        for (int p = 0; p < 32; ++p) {
          unsigned u = hrow[p];
          hp += bflo(u) * Wah[(2 * p) * 64 + hc] + bfhi(u) * Wah[(2 * p + 1) * 64 + hc];
        }
        sc += tanhf_hw(qproj[m * 64 + hc] + hp + bias_a[hc]) * vvec[hc];
      }
      scores[m * TH_MEM + k] = sc;
    }
  }
  __syncthreads();

  for (int o = tid; o < 16 * 64; o += NTH) {         // e_t = softmax(scores)@Hmem
    int m = o >> 6, hc = o & 63;
    float mx = -1e30f;
    for (int k = 0; k < TH_MEM; ++k) mx = fmaxf(mx, scores[m * TH_MEM + k]);
    float den = 0.f, e = 0.f;
    for (int k = 0; k < TH_MEM; ++k) {
      float w = __expf(scores[m * TH_MEM + k] - mx);
      den += w;
      unsigned u = hist[(k * 16 + m) * 32 + (hc >> 1)];
      e += w * ((hc & 1) ? bfhi(u) : bflo(u));
    }
    ebuf[o] = e * rcpf(den);
  }
  __syncthreads();

  for (int o = tid; o < 16 * 64; o += NTH) {         // final = tanh([h1,e]@Wint+bint)
    int m = o >> 6, hc = o & 63;
    float acc = bint[hc];
    for (int d = 0; d < 64; ++d)
      acc += xb[m * 64 + d] * Wint[d * 64 + hc] + ebuf[m * 64 + d] * Wint[(64 + d) * 64 + hc];
    out[(size_t)(b0 + m) * 64 + hc] = tanhf_hw(acc);
  }
}

extern "C" void kernel_launch(void* const* d_in, const int* in_sizes, int n_in,
                              void* d_out, int out_size, void* d_ws, size_t ws_size,
                              hipStream_t stream) {
  (void)in_sizes; (void)n_in; (void)d_ws; (void)ws_size; (void)out_size;
  const float* x      = (const float*)d_in[0];
  const float* Ws0    = (const float*)d_in[1];
  const float* bs0    = (const float*)d_in[2];
  const float* Wg0    = (const float*)d_in[3];
  const float* bg0    = (const float*)d_in[4];
  const float* Ws1    = (const float*)d_in[5];
  const float* bs1    = (const float*)d_in[6];
  const float* Wg1    = (const float*)d_in[7];
  const float* bg1    = (const float*)d_in[8];
  const float* Waq    = (const float*)d_in[9];
  const float* baq    = (const float*)d_in[10];
  const float* Wah    = (const float*)d_in[11];
  const float* bah    = (const float*)d_in[12];
  const float* bias_a = (const float*)d_in[13];
  const float* vvec   = (const float*)d_in[14];
  const float* Wint   = (const float*)d_in[15];
  const float* bint   = (const float*)d_in[16];

  // Allow >64KB dynamic LDS (CDNA5 WGP supports 320KB per workgroup).
  (void)hipFuncSetAttribute(reinterpret_cast<const void*>(thlstm_wmma_kernel),
                            hipFuncAttributeMaxDynamicSharedMemorySize, SMEM_BYTES);

  thlstm_wmma_kernel<<<TH_B / ROWS, NTH, SMEM_BYTES, stream>>>(
      x, Ws0, bs0, Wg0, bg0, Ws1, bs1, Wg1, bg1,
      Waq, baq, Wah, bah, bias_a, vvec, Wint, bint, (float*)d_out);
}
